// EISANImodel_90048284328142
// MI455X (gfx1250) — matrix operand order repaired and checked by reading, not compiled, not run on hardware
//
#include <hip/hip_runtime.h>
#include <hip/hip_bf16.h>

typedef __attribute__((ext_vector_type(16))) __bf16 v16bf;
typedef __attribute__((ext_vector_type(8)))  __bf16 v8bf;
typedef __attribute__((ext_vector_type(8)))  float  v8f;
typedef __attribute__((ext_vector_type(4)))  float  f4;

#define B_    64
#define F_ENC 2048
#define H_    8192
#define C_    100
#define THRF  3.5f   // threshold is 4; sums are exact small integers in f32

// ---------------------------------------------------------------------------
// Swizzled activation layout: WMMA bf16 B operand in this code's contiguous k order (k = 16*half + elem).
// For K-chunk kc (32 rows) and N-chunk nt (16 cols), 512 bf16 are contiguous:
//   lane  = ((k>>4)&1)*16 + (n&15)    (lanes 0-15: K 0-15, lanes 16-31: K 16-31)
//   elem  = k & 15                    (2 halves per VGPR, 8 VGPRs = v16bf)
// so a lane's B tile is ONE contiguous 32-byte load.
// ---------------------------------------------------------------------------
__device__ __forceinline__ size_t swz_idx(int k, int n) {
  return ((size_t)((k >> 5) * (B_ / 16) + (n >> 4)) << 9)
       + (size_t)(((((k >> 4) & 1) << 4) + (n & 15)) * 16 + (k & 15));
}

// ---------------------------------------------------------------------------
// Binarize encoded input x[B][F] -> swizzled bf16 activations (exact 0/1).
// ---------------------------------------------------------------------------
__global__ void binarize_kernel(const float* __restrict__ x,
                                __bf16* __restrict__ a0) {
  int t = blockIdx.x * blockDim.x + threadIdx.x;   // t < F_ENC*B_
  int b = t & (B_ - 1);
  int f = t >> 6;
  float v = x[(size_t)b * F_ENC + f];
  a0[swz_idx(f, b)] = (v > 0.5f) ? (__bf16)1.0f : (__bf16)0.0f;
}

__device__ __forceinline__ void pack8(v16bf& a, int base, f4 lo, f4 hi) {
  a[base + 0] = (__bf16)lo[0]; a[base + 1] = (__bf16)lo[1];
  a[base + 2] = (__bf16)lo[2]; a[base + 3] = (__bf16)lo[3];
  a[base + 4] = (__bf16)hi[0]; a[base + 5] = (__bf16)hi[1];
  a[base + 6] = (__bf16)hi[2]; a[base + 7] = (__bf16)hi[3];
}

// ---------------------------------------------------------------------------
// One layer: z0 = a@Ws0^T, z1 = a@Ws1^T (bf16 WMMA, f32 accum, exact),
// fired = (z0>=4)|(z1>=4).
// One wave owns a single 16x16 (m-tile x n-tile) output block for BOTH
// segments. The 4 waves of one m-tile (nt = 0..3) live in the same workgroup,
// so their shared W rows dedupe in the WGP$/L2: HBM traffic stays at one pass
// over W (~1.15 GB total across layers -> ~50 us at 23.3 TB/s).
// 2048 waves / 256 workgroups per layer, low VGPR pressure for deep MLP.
// A operand (16x32 bf16): lane m=lane&15, K-slice koffA=(lane>>4)*16 -> rows of
// W streamed as 4x float4/lane, converted to bf16 (exact for {-1,0,1}).
// ---------------------------------------------------------------------------
__global__ __launch_bounds__(256) void layer_kernel(
    const float* __restrict__ Ws0, const float* __restrict__ Ws1,
    const __bf16* __restrict__ ain,
    __bf16* __restrict__ aout_swz, __bf16* __restrict__ aout_plain,
    int Fin) {
  const int lane  = threadIdx.x & 31;
  const int wave  = (blockIdx.x * blockDim.x + threadIdx.x) >> 5;
  const int m0    = (wave >> 2) * 16;          // 512 m-tiles cover H=8192
  const int nt    = wave & 3;                  // batch tile 0..3
  const int row   = m0 + (lane & 15);
  const int koffA = (lane >> 4) * 16;          // lanes 0-15: K 0-15, 16-31: K 16-31

  const float* __restrict__ w0 = Ws0 + (size_t)row * Fin;
  const float* __restrict__ w1 = Ws1 + (size_t)row * Fin;
  const __bf16* __restrict__ bbase = ain + ((size_t)nt << 9) + lane * 16;

  v8f acc0 = (v8f)0.0f, acc1 = (v8f)0.0f;

#pragma unroll 2
  for (int k0 = 0; k0 < Fin; k0 += 32) {
    // speculative stream-ahead of W (OOB prefetch is dropped by HW)
    __builtin_prefetch(w0 + k0 + 256, 0, 3);
    __builtin_prefetch(w1 + k0 + 256, 0, 3);

    // ---- B tile: one contiguous 32B load per lane ----
    const v16bf bv =
        *(const v16bf*)(bbase + ((size_t)((k0 >> 5) * (B_ / 16)) << 9));

    // ---- A tiles: 16 fp32 per lane per segment, packed to v16bf ----
    f4 p0 = *(const f4*)(w0 + k0 + koffA);
    f4 p1 = *(const f4*)(w0 + k0 + koffA + 4);
    f4 p2 = *(const f4*)(w0 + k0 + koffA + 8);
    f4 p3 = *(const f4*)(w0 + k0 + koffA + 12);
    f4 q0 = *(const f4*)(w1 + k0 + koffA);
    f4 q1 = *(const f4*)(w1 + k0 + koffA + 4);
    f4 q2 = *(const f4*)(w1 + k0 + koffA + 8);
    f4 q3 = *(const f4*)(w1 + k0 + koffA + 12);
    v16bf av0, av1;
    pack8(av0, 0, p0, p1); pack8(av0, 8, p2, p3);
    pack8(av1, 0, q0, q1); pack8(av1, 8, q2, q3);

    acc0 = __builtin_amdgcn_wmma_f32_16x16x32_bf16(
        false, av0, false, bv, (short)0, acc0, false, false);
    acc1 = __builtin_amdgcn_wmma_f32_16x16x32_bf16(
        false, av1, false, bv, (short)0, acc1, false, false);
  }

  // ---- epilogue: OR of segment thresholds, write swizzled + plain ----
  // D layout: VGPR r holds (m = m0 + r + 8*(lane>=16), n = nt*16 + lane&15)
  const int mShift = (lane >> 4) << 3;         // 0 or 8
  const int lanep  = ((m0 & 16) ? 16 : 0) + (lane & 15);
  v8bf fired;
#pragma unroll
  for (int r = 0; r < 8; ++r) {
    bool f = (acc0[r] >= THRF) || (acc1[r] >= THRF);
    fired[r] = f ? (__bf16)1.0f : (__bf16)0.0f;
  }
  // swizzled store: 8 contiguous bf16 = one b128 per lane
  size_t sidx = ((size_t)((m0 >> 5) * (B_ / 16) + nt) << 9)
              + (size_t)(lanep * 16 + mShift);
  *(v8bf*)(aout_swz + sidx) = fired;
  // plain [H][B] copy for the logits reduction
#pragma unroll
  for (int r = 0; r < 8; ++r) {
    int m = m0 + mShift + r;
    aout_plain[(size_t)m * B_ + nt * 16 + (lane & 15)] = fired[r];
  }
}

// ---------------------------------------------------------------------------
// logits = a1@O0 + a2@O1 + a3@O2  (64x100, ~0.3 GFLOP -> trivial)
// ---------------------------------------------------------------------------
__global__ void zero_kernel(float* __restrict__ out, int n) {
  int t = blockIdx.x * blockDim.x + threadIdx.x;
  if (t < n) out[t] = 0.0f;
}

#define HCHUNK 512
__global__ __launch_bounds__(256) void logits_kernel(
    const __bf16* __restrict__ a1, const __bf16* __restrict__ a2,
    const __bf16* __restrict__ a3,
    const float* __restrict__ O0, const float* __restrict__ O1,
    const float* __restrict__ O2, float* __restrict__ out) {
  const int nchunk = H_ / HCHUNK;
  const int l  = blockIdx.x / nchunk;
  const int h0 = (blockIdx.x % nchunk) * HCHUNK;
  const __bf16* a = (l == 0) ? a1 : (l == 1) ? a2 : a3;
  const float*  O = (l == 0) ? O0 : (l == 1) ? O1 : O2;
  for (int p = threadIdx.x; p < B_ * C_; p += blockDim.x) {
    int b = p / C_, c = p % C_;
    float sum = 0.0f;
    for (int h = h0; h < h0 + HCHUNK; ++h)
      sum += (float)a[(size_t)h * B_ + b] * O[(size_t)h * C_ + c];
    atomicAdd(&out[p], sum);
  }
}

// ---------------------------------------------------------------------------
extern "C" void kernel_launch(void* const* d_in, const int* in_sizes, int n_in,
                              void* d_out, int out_size, void* d_ws,
                              size_t ws_size, hipStream_t stream) {
  const float* x    = (const float*)d_in[0];
  const float* W0s0 = (const float*)d_in[1];
  const float* W0s1 = (const float*)d_in[2];
  const float* W1s0 = (const float*)d_in[3];
  const float* W1s1 = (const float*)d_in[4];
  const float* W2s0 = (const float*)d_in[5];
  const float* W2s1 = (const float*)d_in[6];
  const float* O0   = (const float*)d_in[7];
  const float* O1   = (const float*)d_in[8];
  const float* O2   = (const float*)d_in[9];
  float* out = (float*)d_out;

  __bf16* ws  = (__bf16*)d_ws;                 // ~6.6 MB of bf16 scratch
  __bf16* A0s = ws;                            // F*B   swizzled
  __bf16* A1s = A0s + (size_t)F_ENC * B_;      // H*B   swizzled
  __bf16* A1p = A1s + (size_t)H_ * B_;         // H*B   plain
  __bf16* A2s = A1p + (size_t)H_ * B_;
  __bf16* A2p = A2s + (size_t)H_ * B_;
  __bf16* A3s = A2p + (size_t)H_ * B_;
  __bf16* A3p = A3s + (size_t)H_ * B_;

  // 2048 waves (512 m-tiles x 4 n-tiles), 8 waves per block:
  // waves 8b..8b+7 = m-tiles {2b, 2b+1} x nt {0..3}  -> W rows shared in-WGP.
  const int layer_blocks = (H_ / 16) * 4 / 8;  // 256

  binarize_kernel<<<(F_ENC * B_) / 256, 256, 0, stream>>>(x, A0s);
  layer_kernel<<<layer_blocks, 256, 0, stream>>>(W0s0, W0s1, A0s, A1s, A1p, F_ENC);
  layer_kernel<<<layer_blocks, 256, 0, stream>>>(W1s0, W1s1, A1s, A2s, A2p, H_);
  layer_kernel<<<layer_blocks, 256, 0, stream>>>(W2s0, W2s1, A2s, A3s, A3p, H_);
  zero_kernel<<<(B_ * C_ + 255) / 256, 256, 0, stream>>>(out, B_ * C_);
  logits_kernel<<<3 * (H_ / HCHUNK), 256, 0, stream>>>(A1p, A2p, A3p, O0, O1, O2, out);
}
